// RobustMambaBackbone_30159260352530
// MI455X (gfx1250) — matrix operand order repaired and checked by reading, not compiled, not run on hardware
//
#include <hip/hip_runtime.h>
#include <math.h>

// ---------------- model constants ----------------
#define D_MODEL   1024
#define D_INNER   2048
#define D_STATE   16
#define DT_RANK   64
#define N_LAYERS  24
#define L_TOK     577            // 1 cls + 24*24 patches
#define BATCH     2
#define M_REAL    (BATCH * L_TOK)   // 1154 rows
#define M_PAD     1184              // 37 * 32  (multiple of 32 for 2 M-subtiles/wave)

typedef __attribute__((ext_vector_type(16))) __bf16   v16bf;
typedef __attribute__((ext_vector_type(8)))  float    v8f;
typedef __attribute__((ext_vector_type(8)))  unsigned int v8u;

__device__ __forceinline__ unsigned short f2bf(float f) {
  unsigned int u = __float_as_uint(f);
  unsigned int r = (u + 0x7fffu + ((u >> 16) & 1u)) >> 16;
  return (unsigned short)r;
}

__device__ __forceinline__ v16bf load_bf16x16b(const char* p0, const char* p1) {
  const uint4 lo = *reinterpret_cast<const uint4*>(p0);
  const uint4 hi = *reinterpret_cast<const uint4*>(p1);
  v8u pk = {lo.x, lo.y, lo.z, lo.w, hi.x, hi.y, hi.z, hi.w};
  return __builtin_bit_cast(v16bf, pk);
}

// =====================================================================
// WMMA GEMM:  C[M_PAD x N] (f32) = A[M_PAD x K] (bf16) * W[N x K]^T (bf16)
// One wave per 32 x (NT*16) output tile: 2 M-subtiles x NT N-subtiles,
// 2*NT v_wmma_f32_16x16x32_bf16 per 32-wide K slice. Ping-pong double
// buffering (K loop unrolled by two slices) -> no rotation moves.
// Addresses are formed as uniform_base + 32-bit *byte* offset so loads
// lower to the SADDR + 32-bit-voffset global form; the paired loads and
// prefetches use instruction immediate offsets. Requires K % 64 == 0.
// =====================================================================
template <int NT>
struct KSlice { v16bf a0, a1, b[NT]; };

template <int NT>
__device__ __forceinline__ void load_slice(KSlice<NT>& s,
    const char* __restrict__ Ab, const char* __restrict__ Wb,
    unsigned aboff0, unsigned aboff1, const unsigned* wboff, unsigned kb) {
  // A 16x32 layout: lanes 0-15 hold K {0..7,16..23}; lanes 16-31 {8..15,24..31}
  // (second 8-element group is +16 elements = +32 bytes)
  s.a0 = load_bf16x16b(Ab + (aboff0 + kb), Ab + (aboff0 + kb) + 32);
  s.a1 = load_bf16x16b(Ab + (aboff1 + kb), Ab + (aboff1 + kb) + 32);
  __builtin_prefetch(Ab + (aboff0 + kb) + 128, 0, 3);
  __builtin_prefetch(Ab + (aboff1 + kb) + 128, 0, 3);
  #pragma unroll
  for (int t = 0; t < NT; ++t) {
    // B 32x16 layout: lanes 0-15 hold K {0..7,16..23} of column lm; lanes 16-31 {8..15,24..31}
    // (second 8-element group is +16 elements = +32 bytes)
    s.b[t] = load_bf16x16b(Wb + (wboff[t] + kb), Wb + (wboff[t] + kb) + 32);
    __builtin_prefetch(Wb + (wboff[t] + kb) + 128, 0, 3);
  }
}

template <int NT>
__device__ __forceinline__ void mma_slice(const KSlice<NT>& s, v8f (&acc)[2][NT]) {
  #pragma unroll
  for (int t = 0; t < NT; ++t) {
    acc[0][t] = __builtin_amdgcn_wmma_f32_16x16x32_bf16(
        false, s.a0, false, s.b[t], (short)0, acc[0][t], false, false);
    acc[1][t] = __builtin_amdgcn_wmma_f32_16x16x32_bf16(
        false, s.a1, false, s.b[t], (short)0, acc[1][t], false, false);
  }
}

template <int NT>
__global__ __launch_bounds__(32) void wmma_gemm_bf16(
    const unsigned short* __restrict__ A,
    const unsigned short* __restrict__ W,
    float* __restrict__ C, int N, int K) {
  const int lane = threadIdx.x;          // 0..31
  const int half = lane >> 4;            // 0 or 1
  const int lm   = lane & 15;
  const int col0 = blockIdx.x * (NT * 16);
  const int row0 = blockIdx.y * 32;

  const char* Ab = (const char*)A;
  const char* Wb = (const char*)W;
  // 32-bit per-lane BYTE offsets (staging buffers < 16 MB, always fit)
  const unsigned aboff0 = ((unsigned)(row0 + lm) * (unsigned)K + half * 8) * 2u;
  const unsigned aboff1 = ((unsigned)(row0 + 16 + lm) * (unsigned)K + half * 8) * 2u;
  unsigned wboff[NT];
  #pragma unroll
  for (int t = 0; t < NT; ++t)
    wboff[t] = ((unsigned)(col0 + t * 16 + lm) * (unsigned)K + half * 8) * 2u;

  v8f acc[2][NT];
  #pragma unroll
  for (int mi = 0; mi < 2; ++mi)
    #pragma unroll
    for (int t = 0; t < NT; ++t) acc[mi][t] = (v8f){};

  KSlice<NT> T0, T1;
  load_slice<NT>(T0, Ab, Wb, aboff0, aboff1, wboff, 0u);

  // pairwise-pipelined main loop over 64-byte (32-element) K slices
  const unsigned KB = (unsigned)K * 2u;     // row bytes
  unsigned kb = 64;
  for (; kb + 64 < KB; kb += 128) {
    load_slice<NT>(T1, Ab, Wb, aboff0, aboff1, wboff, kb);
    mma_slice<NT>(T0, acc);
    load_slice<NT>(T0, Ab, Wb, aboff0, aboff1, wboff, kb + 64);
    mma_slice<NT>(T1, acc);
  }
  // epilogue: last odd slice (kb == KB - 64) then drain both buffers
  load_slice<NT>(T1, Ab, Wb, aboff0, aboff1, wboff, kb);
  mma_slice<NT>(T0, acc);
  mma_slice<NT>(T1, acc);

  // C 16x16 f32 layout: lane lm = column, VGPR v = row (half selects rows 0-7/8-15)
  #pragma unroll
  for (int mi = 0; mi < 2; ++mi) {
    float* crow = C + (size_t)(row0 + mi * 16 + half * 8) * N + col0 + lm;
    #pragma unroll
    for (int t = 0; t < NT; ++t)
      #pragma unroll
      for (int v = 0; v < 8; ++v) crow[(size_t)v * N + t * 16] = acc[mi][t][v];
  }
}

// ---------------- f32 -> bf16 weight cast ----------------
__global__ void cast_w_kernel(const float* __restrict__ src,
                              unsigned short* __restrict__ dst, int n) {
  int i = blockIdx.x * 256 + threadIdx.x;
  if (i < n) dst[i] = f2bf(src[i]);
}

// ---------------- LayerNorm (D=1024) + bf16 cast, zero pad rows ----------------
__global__ void ln_cast_kernel(const float* __restrict__ x,
                               const float* __restrict__ g,
                               const float* __restrict__ bb,
                               unsigned short* __restrict__ out, int Mreal) {
  int row = blockIdx.x;
  int t = threadIdx.x;
  __shared__ float s1[256], s2[256];
  if (row >= Mreal) {
    for (int c = t; c < D_MODEL; c += 256) out[(size_t)row * D_MODEL + c] = 0;
    return;
  }
  const float* xr = x + (size_t)row * D_MODEL;
  float v[4];
  float sum = 0.f;
  #pragma unroll
  for (int i = 0; i < 4; ++i) { v[i] = xr[t + i * 256]; sum += v[i]; }
  s1[t] = sum; __syncthreads();
  for (int st = 128; st > 0; st >>= 1) { if (t < st) s1[t] += s1[t + st]; __syncthreads(); }
  float mean = s1[0] * (1.f / D_MODEL);
  float sq = 0.f;
  #pragma unroll
  for (int i = 0; i < 4; ++i) { float d = v[i] - mean; sq += d * d; }
  s2[t] = sq; __syncthreads();
  for (int st = 128; st > 0; st >>= 1) { if (t < st) s2[t] += s2[t + st]; __syncthreads(); }
  float rstd = rsqrtf(s2[0] * (1.f / D_MODEL) + 1e-5f);
  #pragma unroll
  for (int i = 0; i < 4; ++i) {
    int c = t + i * 256;
    out[(size_t)row * D_MODEL + c] = f2bf((v[i] - mean) * rstd * g[c] + bb[c]);
  }
}

// ---------------- depthwise conv4 (causal) + SiLU ----------------
__global__ void conv_silu_kernel(const float* __restrict__ xz,
                                 const float* __restrict__ dcw,
                                 const float* __restrict__ dcb,
                                 float* __restrict__ xc,
                                 unsigned short* __restrict__ xcb, int Mreal) {
  size_t idx = (size_t)blockIdx.x * 256 + threadIdx.x;
  if (idx >= (size_t)M_PAD * D_INNER) return;
  int d = (int)(idx % D_INNER);
  int row = (int)(idx / D_INNER);
  if (row >= Mreal) { xcb[idx] = 0; return; }
  int b = row / L_TOK, l = row % L_TOK;
  float acc = dcb[d];
  #pragma unroll
  for (int k = 0; k < 4; ++k) {
    int ll = l + k - 3;
    if (ll >= 0)
      acc += xz[((size_t)(b * L_TOK + ll)) * (2 * D_INNER) + d] * dcw[d * 4 + k];
  }
  float s = acc / (1.f + __expf(-acc));
  xc[idx] = s;
  xcb[idx] = f2bf(s);
}

// ---------------- slice dbc[:, :64] -> bf16 ----------------
__global__ void slice64_cast_kernel(const float* __restrict__ dbc,
                                    unsigned short* __restrict__ out, int Mreal) {
  int idx = blockIdx.x * 256 + threadIdx.x;
  if (idx >= M_PAD * DT_RANK) return;
  int row = idx / DT_RANK, c = idx % DT_RANK;
  out[idx] = (row < Mreal) ? f2bf(dbc[(size_t)row * 96 + c]) : (unsigned short)0;
}

// ---------------- dt = softplus(dtraw + dtb) in place ----------------
__global__ void softplus_bias_kernel(float* __restrict__ dt,
                                     const float* __restrict__ dtb, int Mreal) {
  size_t idx = (size_t)blockIdx.x * 256 + threadIdx.x;
  if (idx >= (size_t)Mreal * D_INNER) return;
  int d = (int)(idx % D_INNER);
  float v = dt[idx] + dtb[d];
  dt[idx] = (v > 20.f) ? v : log1pf(__expf(v));
}

// ---------------- selective scan: one thread per (b,d) chain ----------------
__global__ void scan_kernel(const float* __restrict__ dt,
                            const float* __restrict__ xc,
                            const float* __restrict__ dbc,
                            const float* __restrict__ xz,
                            const float* __restrict__ A_log,
                            const float* __restrict__ Dp,
                            unsigned short* __restrict__ yb) {
  int tid = blockIdx.x * 256 + threadIdx.x;     // 0 .. 4095
  if (tid >= BATCH * D_INNER) return;
  int d = tid & (D_INNER - 1);
  int b = tid >> 11;
  float a[D_STATE], h[D_STATE];
  #pragma unroll
  for (int s = 0; s < D_STATE; ++s) {
    a[s] = -__expf(A_log[d * D_STATE + s]);
    h[s] = 0.f;
  }
  const float dpv = Dp[d];
  for (int l = 0; l < L_TOK; ++l) {
    size_t row = (size_t)b * L_TOK + l;
    float dtv = dt[row * D_INNER + d];
    float xv  = xc[row * D_INNER + d];
    const float* dr = dbc + row * 96;
    float dtx = dtv * xv;
    float y = 0.f;
    #pragma unroll
    for (int s = 0; s < D_STATE; ++s) {
      h[s] = __expf(dtv * a[s]) * h[s] + dtx * dr[DT_RANK + s];
      y += h[s] * dr[DT_RANK + D_STATE + s];
    }
    y += xv * dpv;
    float z = xz[row * (2 * D_INNER) + D_INNER + d];
    y *= z / (1.f + __expf(-z));
    yb[row * D_INNER + d] = f2bf(y);
  }
}

// ---------------- zero pad rows of a bf16 buffer ----------------
__global__ void zero_rows_kernel(unsigned short* __restrict__ buf,
                                 int startRow, int endRow, int cols) {
  int idx = blockIdx.x * 256 + threadIdx.x;
  int total = (endRow - startRow) * cols;
  if (idx >= total) return;
  buf[(size_t)(startRow + idx / cols) * cols + (idx % cols)] = 0;
}

// ---------------- GELU(x + bias) -> bf16, zero pad rows ----------------
__global__ void gelu_bias_cast_kernel(const float* __restrict__ h1,
                                      const float* __restrict__ f1b,
                                      unsigned short* __restrict__ out, int Mreal) {
  size_t idx = (size_t)blockIdx.x * 256 + threadIdx.x;
  if (idx >= (size_t)M_PAD * (4 * D_MODEL)) return;
  int row = (int)(idx / (4 * D_MODEL));
  int c   = (int)(idx % (4 * D_MODEL));
  if (row >= Mreal) { out[idx] = 0; return; }
  float v = h1[idx] + f1b[c];
  float g = 0.5f * v * (1.f + erff(v * 0.70710678118654752f));
  out[idx] = f2bf(g);
}

// ---------------- tokens += h2 + f2b ----------------
__global__ void residual_kernel(float* __restrict__ tokens,
                                const float* __restrict__ h2,
                                const float* __restrict__ f2b, int Mreal) {
  size_t idx = (size_t)blockIdx.x * 256 + threadIdx.x;
  if (idx >= (size_t)Mreal * D_MODEL) return;
  int d = (int)(idx % D_MODEL);
  tokens[idx] += h2[idx] + f2b[d];
}

// ---------------- patch embed + cls + sincos PE ----------------
__global__ void patch_embed_kernel(const float* __restrict__ x,
                                   const float* __restrict__ pe_w,
                                   const float* __restrict__ pe_b,
                                   const float* __restrict__ cls,
                                   float* __restrict__ tokens) {
  size_t idx = (size_t)blockIdx.x * 256 + threadIdx.x;
  if (idx >= (size_t)M_REAL * D_MODEL) return;
  int d = (int)(idx % D_MODEL);
  int row = (int)(idx / D_MODEL);
  int b = row / L_TOK, t = row % L_TOK;
  float val;
  if (t == 0) {
    val = cls[d];                       // pe row 0 is zero
  } else {
    int hp = (t - 1) / 24, wp = (t - 1) % 24;
    val = pe_b[d];
    const float* wptr = pe_w + (size_t)d * 192;     // [3][8][8]
    const float* xb = x + (size_t)b * 3 * 192 * 192;
    #pragma unroll
    for (int c = 0; c < 3; ++c)
      #pragma unroll
      for (int p = 0; p < 8; ++p) {
        const float* xr = xb + ((size_t)c * 192 + hp * 8 + p) * 192 + wp * 8;
        const float* wr = wptr + (c * 8 + p) * 8;
        #pragma unroll
        for (int q = 0; q < 8; ++q) val += xr[q] * wr[q];
      }
    // 2D sincos PE: dims [0,512) from hp, [512,1024) from wp
    float pos; int j; bool use_cos;
    if (d < 512) { pos = (float)hp; j = d & 255; use_cos = (d >= 256); }
    else { int e = d - 512; pos = (float)wp; j = e & 255; use_cos = (e >= 256); }
    float inv = __expf(-(float)j * 0.03597789203877064f);   // ln(1e4)/256
    float ang = pos * inv;
    val += use_cos ? __cosf(ang) : __sinf(ang);
  }
  tokens[idx] = val;
}

// =====================================================================
// Host-side orchestration
// =====================================================================
extern "C" void kernel_launch(void* const* d_in, const int* in_sizes, int n_in,
                              void* d_out, int out_size, void* d_ws, size_t ws_size,
                              hipStream_t stream) {
  (void)in_sizes; (void)n_in; (void)out_size; (void)ws_size;

  const float* x     = (const float*)d_in[0];
  const float* pe_w  = (const float*)d_in[1];
  const float* pe_b  = (const float*)d_in[2];
  const float* cls   = (const float*)d_in[3];
  const float* g1    = (const float*)d_in[4];
  const float* b1    = (const float*)d_in[5];
  const float* in_w  = (const float*)d_in[6];
  const float* dcw   = (const float*)d_in[7];
  const float* dcb   = (const float*)d_in[8];
  const float* xpw   = (const float*)d_in[9];
  const float* dtw   = (const float*)d_in[10];
  const float* dtb   = (const float*)d_in[11];
  const float* A_log = (const float*)d_in[12];
  const float* Dp    = (const float*)d_in[13];
  const float* ow    = (const float*)d_in[14];
  const float* g2    = (const float*)d_in[15];
  const float* b2    = (const float*)d_in[16];
  const float* f1w   = (const float*)d_in[17];
  const float* f1b   = (const float*)d_in[18];
  const float* f2w   = (const float*)d_in[19];
  const float* f2b   = (const float*)d_in[20];

  float* tokens = (float*)d_out;   // [M_REAL x 1024] rows are contiguous == output

  // workspace carve-up (256B aligned)
  char* base = (char*)d_ws;
  size_t o = 0;
  auto carve = [&](size_t bytes) { size_t r = o; o += (bytes + 255) & ~(size_t)255; return r; };
  unsigned short* actIn = (unsigned short*)(base + carve((size_t)M_PAD * 4096 * 2));
  unsigned short* wbuf  = (unsigned short*)(base + carve((size_t)4096 * 1024 * 2));
  float* xzb = (float*)(base + carve((size_t)M_PAD * 4096 * 4));
  float* xcb = (float*)(base + carve((size_t)M_PAD * 2048 * 4));
  float* dbc = (float*)(base + carve((size_t)M_PAD * 96 * 4));
  float* dtv = (float*)(base + carve((size_t)M_PAD * 2048 * 4));
  float* mo  = (float*)(base + carve((size_t)M_PAD * 1024 * 4));
  float* h1  = (float*)(base + carve((size_t)M_PAD * 4096 * 4));
  float* h2  = (float*)(base + carve((size_t)M_PAD * 1024 * 4));

  const int MT = M_PAD / 32;   // 37 row-tiles (32 rows / wave)
  auto ceil256 = [](size_t n) { return (unsigned)((n + 255) / 256); };

  patch_embed_kernel<<<ceil256((size_t)M_REAL * 1024), 256, 0, stream>>>(
      x, pe_w, pe_b, cls, tokens);

  for (int L = 0; L < N_LAYERS; ++L) {
    const float* g1l  = g1  + (size_t)L * 1024;
    const float* b1l  = b1  + (size_t)L * 1024;
    const float* inwl = in_w+ (size_t)L * 4096 * 1024;
    const float* dcwl = dcw + (size_t)L * 2048 * 4;
    const float* dcbl = dcb + (size_t)L * 2048;
    const float* xpwl = xpw + (size_t)L * 96 * 2048;
    const float* dtwl = dtw + (size_t)L * 2048 * 64;
    const float* dtbl = dtb + (size_t)L * 2048;
    const float* Alog = A_log+(size_t)L * 2048 * 16;
    const float* Dpl  = Dp  + (size_t)L * 2048;
    const float* owl  = ow  + (size_t)L * 1024 * 2048;
    const float* g2l  = g2  + (size_t)L * 1024;
    const float* b2l  = b2  + (size_t)L * 1024;
    const float* f1wl = f1w + (size_t)L * 4096 * 1024;
    const float* f1bl = f1b + (size_t)L * 4096;
    const float* f2wl = f2w + (size_t)L * 1024 * 4096;
    const float* f2bl = f2b + (size_t)L * 1024;

    // LN1 -> bf16
    ln_cast_kernel<<<M_PAD, 256, 0, stream>>>(tokens, g1l, b1l, actIn, M_REAL);
    // xz = a1 @ in_w^T   [Mp x 4096]
    cast_w_kernel<<<ceil256(4096 * 1024), 256, 0, stream>>>(inwl, wbuf, 4096 * 1024);
    wmma_gemm_bf16<4><<<dim3(4096 / 64, MT), 32, 0, stream>>>(actIn, wbuf, xzb, 4096, 1024);
    // conv4 + SiLU -> xc (f32) and bf16 operand
    conv_silu_kernel<<<ceil256((size_t)M_PAD * 2048), 256, 0, stream>>>(
        xzb, dcwl, dcbl, xcb, actIn, M_REAL);
    // dbc = xc @ xpw^T   [Mp x 96]
    cast_w_kernel<<<ceil256(96 * 2048), 256, 0, stream>>>(xpwl, wbuf, 96 * 2048);
    wmma_gemm_bf16<2><<<dim3(96 / 32, MT), 32, 0, stream>>>(actIn, wbuf, dbc, 96, 2048);
    // dt = softplus(dbc[:, :64] @ dtw^T + dtb)
    slice64_cast_kernel<<<ceil256((size_t)M_PAD * 64), 256, 0, stream>>>(dbc, actIn, M_REAL);
    cast_w_kernel<<<ceil256(2048 * 64), 256, 0, stream>>>(dtwl, wbuf, 2048 * 64);
    wmma_gemm_bf16<4><<<dim3(2048 / 64, MT), 32, 0, stream>>>(actIn, wbuf, dtv, 2048, 64);
    softplus_bias_kernel<<<ceil256((size_t)M_REAL * 2048), 256, 0, stream>>>(dtv, dtbl, M_REAL);
    // selective scan -> y (silu-gated) as bf16 operand
    scan_kernel<<<ceil256(BATCH * D_INNER), 256, 0, stream>>>(
        dtv, xcb, dbc, xzb, Alog, Dpl, actIn);
    zero_rows_kernel<<<ceil256((size_t)(M_PAD - M_REAL) * 2048), 256, 0, stream>>>(
        actIn, M_REAL, M_PAD, 2048);
    // mamba out = y @ ow^T   [Mp x 1024]
    cast_w_kernel<<<ceil256(1024 * 2048), 256, 0, stream>>>(owl, wbuf, 1024 * 2048);
    wmma_gemm_bf16<4><<<dim3(1024 / 64, MT), 32, 0, stream>>>(actIn, wbuf, mo, 1024, 2048);
    // LN2 -> bf16
    ln_cast_kernel<<<M_PAD, 256, 0, stream>>>(mo, g2l, b2l, actIn, M_REAL);
    // MLP up  [Mp x 4096]
    cast_w_kernel<<<ceil256(4096 * 1024), 256, 0, stream>>>(f1wl, wbuf, 4096 * 1024);
    wmma_gemm_bf16<4><<<dim3(4096 / 64, MT), 32, 0, stream>>>(actIn, wbuf, h1, 4096, 1024);
    gelu_bias_cast_kernel<<<ceil256((size_t)M_PAD * 4096), 256, 0, stream>>>(
        h1, f1bl, actIn, M_REAL);
    // MLP down [Mp x 1024]
    cast_w_kernel<<<ceil256(1024 * 4096), 256, 0, stream>>>(f2wl, wbuf, 1024 * 4096);
    wmma_gemm_bf16<4><<<dim3(1024 / 64, MT), 32, 0, stream>>>(actIn, wbuf, h2, 1024, 4096);
    // tokens += h2 + f2b
    residual_kernel<<<ceil256((size_t)M_REAL * 1024), 256, 0, stream>>>(
        tokens, h2, f2bl, M_REAL);
  }
}
